// TLSTMModel_32031866093740
// MI455X (gfx1250) — matrix operand order repaired and checked by reading, not compiled, not run on hardware
//
#include <hip/hip_runtime.h>
#include <hip/hip_bf16.h>

// ---------------------------------------------------------------------------
// Time-aware LSTM on MI455X (gfx1250, wave32, WMMA + TDM).
//   B=1024, S=200, F1=129 (F=128 feats + 1 dt), H=256, C=3
// Strategy:
//   prep_kernel   : f32->bf16 weight conversion, bias fusion, h/c zero-init
//   embed_kernel  : emb = tanh(feats @ W_in^T + b_in)  (WMMA bf16, f32 accum)
//   step_kernel   : per time-step, fused gates = emb_s@W_ih^T + h@W_hh^T
//                   + LSTM cell update (gamma decay + mask). Weight B-tiles
//                   staged in LDS by the Tensor Data Mover (one 3-D descriptor
//                   per phase: 4 gates x 16 rows x 512B, LDS pitch padding
//                   done by TDM pad_interval/pad_amount). A operand software-
//                   pipelined from global. h double-buffered.
//   out_kernel    : tiny final projection h @ W_out^T + b_out
// ---------------------------------------------------------------------------

typedef __attribute__((ext_vector_type(16))) __bf16 v16bf;
typedef __attribute__((ext_vector_type(8)))  __bf16 v8bf;
typedef __attribute__((ext_vector_type(8)))  float  v8f;
typedef __attribute__((ext_vector_type(4)))  unsigned int v4u;
typedef __attribute__((ext_vector_type(8)))  int    v8i;
typedef __attribute__((ext_vector_type(4)))  int    v4i;

#define B_   1024
#define S_   200
#define F1_  129
#define F_   128
#define H_   256
#define C_   3
#define G4H_ (4 * H_)

// LDS pitch for staged weight rows: 256 K + 8 pad bf16 = 132 dwords, so
// successive rows are offset 4 banks -> B-fragment reads spread across banks.
#define PITCH_ 264

#if defined(__has_builtin)
#if __has_builtin(__builtin_amdgcn_tensor_load_to_lds) && \
    __has_builtin(__builtin_amdgcn_s_wait_tensorcnt)
#define USE_TDM 1
#endif
#endif
#ifndef USE_TDM
#define USE_TDM 0
#endif

__device__ __forceinline__ float sigmf(float x) { return 1.0f / (1.0f + __expf(-x)); }

// A-operand (16x32 bf16) per-lane load, row-major source with K contiguous.
// Lane L: M = L&15, half = L>>4. elems 0..7 -> K = k0 + 8*half + i,
// elems 8..15 -> K = k0 + 16 + 8*half + i.   (ISA 7.12.2, 16-bit A 16x32)
__device__ __forceinline__ v16bf load_a_bf16(const __bf16* row, int k0, int half) {
  const v8bf lo = *(const v8bf*)(row + k0 + half * 8);
  const v8bf hi = *(const v8bf*)(row + k0 + 16 + half * 8);
  v16bf a;
#pragma unroll
  for (int i = 0; i < 8; ++i) { a[i] = lo[i]; a[i + 8] = hi[i]; }
  return a;
}

__device__ __forceinline__ v16bf load_a_f32(const float* row, int k0, int half) {
  const float* p0 = row + k0 + half * 8;
  const float* p1 = row + k0 + 16 + half * 8;
  v16bf a;
#pragma unroll
  for (int i = 0; i < 8; ++i) { a[i] = (__bf16)p0[i]; a[i + 8] = (__bf16)p1[i]; }
  return a;
}

// B-operand (32x16 bf16): lane (N = L&15) reads K = k0 + 8*half + i (elems 0..7)
// and k0 + 16 + 8*half + i (elems 8..15).  Global variant (row-major weights):
__device__ __forceinline__ v16bf load_b_bf16(const __bf16* nrow, int k0, int half) {
  const v8bf lo = *(const v8bf*)(nrow + k0 + half * 8);
  const v8bf hi = *(const v8bf*)(nrow + k0 + 16 + half * 8);
  v16bf b;
#pragma unroll
  for (int i = 0; i < 8; ++i) { b[i] = lo[i]; b[i + 8] = hi[i]; }
  return b;
}

// LDS variant: two 16-byte reads (ds_load_b128) to respect 16B row alignment.
__device__ __forceinline__ v16bf load_b_lds(const __bf16* p) {
  const v8bf b0 = *(const v8bf*)(p);
  const v8bf b1 = *(const v8bf*)(p + 16);
  v16bf b;
#pragma unroll
  for (int i = 0; i < 8; ++i) { b[i] = b0[i]; b[i + 8] = b1[i]; }
  return b;
}

// ---------------------------------------------------------------------------
// prep: weight conversion to bf16, fused bias, zero h0/c.  i in [0, 262144)
// (= 4H*H = B*H, the largest of the per-element jobs).
// ---------------------------------------------------------------------------
__global__ void prep_kernel(const float* __restrict__ W_in, const float* __restrict__ W_ih,
                            const float* __restrict__ W_hh, const float* __restrict__ b_ih,
                            const float* __restrict__ b_hh,
                            __bf16* __restrict__ win, __bf16* __restrict__ wih,
                            __bf16* __restrict__ whh, float* __restrict__ bsum,
                            __bf16* __restrict__ h0, float* __restrict__ c) {
  const int i = blockIdx.x * blockDim.x + threadIdx.x;
  if (i < H_ * F_) win[i] = (__bf16)W_in[i];
  wih[i] = (__bf16)W_ih[i];
  whh[i] = (__bf16)W_hh[i];
  if (i < G4H_) bsum[i] = b_ih[i] + b_hh[i];
  h0[i] = (__bf16)0.0f;
  c[i]  = 0.0f;
}

// ---------------------------------------------------------------------------
// embed: emb[(b*S+s), h] = tanh(feats @ W_in^T + b_in), bf16 output.
// grid = (12800/8, 16) blocks of 256 threads; wave w -> M-tile blockIdx.x*8+w.
// ---------------------------------------------------------------------------
__global__ void embed_kernel(const float* __restrict__ x, const __bf16* __restrict__ win,
                             const float* __restrict__ b_in, __bf16* __restrict__ emb) {
  const int lane = threadIdx.x & 31;
  const int wave = threadIdx.x >> 5;
  const int half = lane >> 4;
  const int lm   = lane & 15;
  const int m0 = (blockIdx.x * 8 + wave) * 16;   // row tile in [0, 204800)
  const int n0 = blockIdx.y * 16;                // hidden tile in [0, 256)

  const float*  arow = x + (size_t)(m0 + lm) * F1_;      // feats: K = 0..127 (<F1-1)
  const __bf16* brow = win + (size_t)(n0 + lm) * F_;

  v8f acc = {};
#pragma unroll
  for (int k0 = 0; k0 < F_; k0 += 32) {
    v16bf a = load_a_f32(arow, k0, half);
    v16bf b = load_b_bf16(brow, k0, half);
    acc = __builtin_amdgcn_wmma_f32_16x16x32_bf16(false, a, false, b, (short)0, acc,
                                                  false, false);
  }
#pragma unroll
  for (int r = 0; r < 8; ++r) {
    const int m = m0 + r + 8 * half;   // C/D layout: M = r (lanes 0-15), r+8 (16-31)
    const int n = n0 + lm;
    emb[(size_t)m * H_ + n] = (__bf16)tanhf(acc[r] + b_in[n]);
  }
}

// ---------------------------------------------------------------------------
// step: one time step.  grid = (B/128, H/16), 256 threads (8 waves).
// Wave owns a 16(batch) x 16(hidden) tile; 4 accumulators (i,f,g,o) share the
// A operand -> 64 WMMAs/wave over K=512 ([emb_s | h_in] x [W_ih | W_hh]).
// Weight B-tiles (4 gates x 16 rows x 256 K bf16 = 32KB/phase) are DMA'd into
// LDS by the TDM (one descriptor per phase), shared by all 8 waves.  The
// element-wise TLSTM update happens entirely in-lane.
// ---------------------------------------------------------------------------
__global__ void step_kernel(const __bf16* __restrict__ emb, const __bf16* __restrict__ h_in,
                            __bf16* __restrict__ h_out, float* __restrict__ c,
                            const __bf16* __restrict__ wih, const __bf16* __restrict__ whh,
                            const float* __restrict__ bsum, const float* __restrict__ x,
                            const int* __restrict__ lengths,
                            const float* __restrict__ W_decay,
                            const float* __restrict__ b_decay, int s) {
  __shared__ __bf16 lw[64 * PITCH_];   // 33 KB: one phase of weight rows

  const int tid  = threadIdx.x;
  const int lane = tid & 31;
  const int wave = tid >> 5;
  const int half = lane >> 4;
  const int lm   = lane & 15;
  const int m0 = blockIdx.x * 128 + wave * 16;   // batch tile
  const int n0 = blockIdx.y * 16;                // hidden tile

  v8f acc[4] = {v8f{}, v8f{}, v8f{}, v8f{}};

  const __bf16* asrc[2] = {emb + ((size_t)(m0 + lm) * S_ + (size_t)s) * H_,
                           h_in + (size_t)(m0 + lm) * H_};
  const __bf16* wsrc[2] = {wih, whh};

  for (int p = 0; p < 2; ++p) {
#if USE_TDM
    // One 3-D TDM descriptor stages this block's 64 weight rows:
    //   x: 128 dwords (one 512B row), y: 16 rows (contiguous: stride == width),
    //   z: 4 gates (tensor_dim1_stride = 256 rows * 128 dwords = 32768).
    // TDM inserts 4 dwords of LDS padding after every 128 dwords, matching
    // PITCH_ = 264 bf16.  Issued by wave 0 only (TDM ignores EXEC).
    if (tid < 32) {
      const unsigned long long ga =
          (unsigned long long)(uintptr_t)(wsrc[p] + (size_t)n0 * H_);
      const unsigned int la = (unsigned int)(uintptr_t)(&lw[0]);
      v4u g0;
      g0[0] = 1u;                                   // count=1, user descriptor
      g0[1] = la;                                   // lds_addr
      g0[2] = (unsigned int)ga;                     // global_addr[31:0]
      g0[3] = (unsigned int)((ga >> 32) & 0x1FFFFFFu) | (2u << 30);  // [56:32]|type=2
      v8i g1;
      g1[0] = (int)((2u << 16)      // data_size = 4B
                    | (1u << 20)    // pad_enable
                    | (6u << 22)    // pad_interval: 128 dwords
                    | (3u << 25));  // pad_amount: 4 dwords
      g1[1] = (int)(128u << 16);    // tensor_dim0 = 128 (dwords per row)
      g1[2] = (int)(16u << 16);     // tensor_dim1 = 16 rows
      g1[3] = (int)(128u << 16);    // tile_dim0 = 128
      g1[4] = (int)(16u | (4u << 16));  // tile_dim1 = 16, tile_dim2 = 4 gates
      g1[5] = 128;                  // tensor_dim0_stride = 128 dwords
      g1[6] = (int)(32768u << 16);  // tensor_dim1_stride = 32768 dwords (lo16)
      g1[7] = 0;                    //   (hi bits zero)
      v4i g2;
      g2[0] = 4;                    // tensor_dim2 = 4 (z extent)
      g2[1] = 0; g2[2] = 0; g2[3] = 0;
      v4i g3;
      g3[0] = 0; g3[1] = 0; g3[2] = 0; g3[3] = 0;
      v8i g4;                       // extra group (6-arg toolchain): unused
      g4[0] = 0; g4[1] = 0; g4[2] = 0; g4[3] = 0;
      g4[4] = 0; g4[5] = 0; g4[6] = 0; g4[7] = 0;
      __builtin_amdgcn_tensor_load_to_lds(g0, g1, g2, g3, g4, 0);
      __builtin_amdgcn_s_wait_tensorcnt(0);
    }
    __syncthreads();
#else
    // Fallback: cooperative VMEM->VGPR->LDS copy of the 64 weight rows.
    {
      const int row = tid >> 2;                       // 0..63
      const int seg = tid & 3;                        // 128B segment
      const int g   = row >> 4;
      const int nn  = n0 + (row & 15);
      const uint4* src = (const uint4*)(wsrc[p] + (size_t)(g * H_ + nn) * H_) + seg * 8;
      uint4*       dst = (uint4*)(lw + row * PITCH_) + seg * 8;
#pragma unroll
      for (int j = 0; j < 8; ++j) dst[j] = src[j];
    }
    __syncthreads();
#endif

    const __bf16* arow = asrc[p];
    v16bf a = load_a_bf16(arow, 0, half);
#pragma unroll
    for (int k0 = 0; k0 < H_; k0 += 32) {
      v16bf an = a;
      if (k0 + 32 < H_) an = load_a_bf16(arow, k0 + 32, half);  // prefetch next
#pragma unroll
      for (int g = 0; g < 4; ++g) {
        v16bf b = load_b_lds(lw + (g * 16 + lm) * PITCH_ + k0 + half * 8);
        acc[g] = __builtin_amdgcn_wmma_f32_16x16x32_bf16(false, a, false, b, (short)0,
                                                         acc[g], false, false);
      }
      a = an;
    }
    __syncthreads();   // protect LDS before next phase overwrites it
  }

  // Element-wise TLSTM update: gates for (b, h) live in the same lane slot
  // across the 4 accumulators (C layout: M = r + 8*half, N = lane&15).
  const int   n  = n0 + lm;
  const float wd = W_decay[n];
  const float bd = b_decay[n];
#pragma unroll
  for (int r = 0; r < 8; ++r) {
    const int m = m0 + r + 8 * half;
    const float gi = acc[0][r] + bsum[0 * H_ + n];
    const float gf = acc[1][r] + bsum[1 * H_ + n];
    const float gg = acc[2][r] + bsum[2 * H_ + n];
    const float go = acc[3][r] + bsum[3 * H_ + n];
    const float dt    = x[(size_t)m * (S_ * F1_) + (size_t)s * F1_ + F_];
    const float gamma = sigmf(dt * wd + bd);
    const float cv = c[(size_t)m * H_ + n] * gamma;
    float cn = sigmf(gf) * cv + sigmf(gi) * tanhf(gg);
    float hn = sigmf(go) * tanhf(cn);
    const float msk = (s < lengths[m]) ? 1.0f : 0.0f;
    cn *= msk;
    hn *= msk;
    c[(size_t)m * H_ + n]     = cn;
    h_out[(size_t)m * H_ + n] = (__bf16)hn;
  }
}

// ---------------------------------------------------------------------------
// out: out[b, c] = h[b, :] . W_out[c, :] + b_out[c]   (1024 x 3, trivial)
// ---------------------------------------------------------------------------
__global__ void out_kernel(const __bf16* __restrict__ h, const float* __restrict__ W_out,
                           const float* __restrict__ b_out, float* __restrict__ out) {
  const int i = blockIdx.x * blockDim.x + threadIdx.x;
  if (i >= B_ * C_) return;
  const int b  = i / C_;
  const int cc = i % C_;
  const __bf16* hr = h + (size_t)b * H_;
  const float*  wr = W_out + (size_t)cc * H_;
  float acc = 0.0f;
#pragma unroll 8
  for (int k = 0; k < H_; ++k) acc += (float)hr[k] * wr[k];
  out[i] = acc + b_out[cc];
}

// ---------------------------------------------------------------------------
extern "C" void kernel_launch(void* const* d_in, const int* in_sizes, int n_in,
                              void* d_out, int out_size, void* d_ws, size_t ws_size,
                              hipStream_t stream) {
  const float* x       = (const float*)d_in[0];
  const int*   lengths = (const int*)d_in[1];
  const float* W_in    = (const float*)d_in[2];
  const float* b_in    = (const float*)d_in[3];
  const float* W_decay = (const float*)d_in[4];
  const float* b_decay = (const float*)d_in[5];
  const float* W_ih    = (const float*)d_in[6];
  const float* W_hh    = (const float*)d_in[7];
  const float* b_ih    = (const float*)d_in[8];
  const float* b_hh    = (const float*)d_in[9];
  const float* W_out   = (const float*)d_in[10];
  const float* b_out   = (const float*)d_in[11];
  float* out = (float*)d_out;

  // Scratch layout (256B aligned), ~108 MB total.
  char* ws = (char*)d_ws;
  size_t off = 0;
  auto take = [&](size_t bytes) {
    void* p = ws + off;
    off = (off + bytes + 255) & ~(size_t)255;
    return p;
  };
  __bf16* emb  = (__bf16*)take((size_t)B_ * S_ * H_ * sizeof(__bf16)); // 104.9 MB
  __bf16* h0   = (__bf16*)take((size_t)B_ * H_ * sizeof(__bf16));
  __bf16* h1   = (__bf16*)take((size_t)B_ * H_ * sizeof(__bf16));
  float*  c    = (float*) take((size_t)B_ * H_ * sizeof(float));
  __bf16* win  = (__bf16*)take((size_t)H_ * F_ * sizeof(__bf16));
  __bf16* wih  = (__bf16*)take((size_t)G4H_ * H_ * sizeof(__bf16));
  __bf16* whh  = (__bf16*)take((size_t)G4H_ * H_ * sizeof(__bf16));
  float*  bsum = (float*) take((size_t)G4H_ * sizeof(float));
  (void)ws_size; (void)in_sizes; (void)n_in; (void)out_size;

  // 1) weights -> bf16, bias fusion, zero h0/c (deterministic per launch)
  prep_kernel<<<(B_ * H_) / 256, 256, 0, stream>>>(W_in, W_ih, W_hh, b_ih, b_hh,
                                                   win, wih, whh, bsum, h0, c);

  // 2) embedding GEMM: 204800 x 128 -> 256, tanh, bf16 out
  embed_kernel<<<dim3((B_ * S_ / 16) / 8, H_ / 16), 256, 0, stream>>>(x, win, b_in, emb);

  // 3) 200 recurrent steps, h double-buffered (h_in read by all hidden tiles)
  __bf16* hb[2] = {h0, h1};
  for (int s = 0; s < S_; ++s) {
    step_kernel<<<dim3(B_ / 128, H_ / 16), 256, 0, stream>>>(
        emb, hb[s & 1], hb[(s + 1) & 1], c, wih, whh, bsum, x, lengths, W_decay,
        b_decay, s);
  }

  // 4) final projection (h after S steps lives in hb[S_ & 1])
  out_kernel<<<(B_ * C_ + 255) / 256, 256, 0, stream>>>(hb[S_ & 1], W_out, b_out, out);
}
